// LocalQMixer_73864847556783
// MI455X (gfx1250) — matrix operand order repaired and checked by reading, not compiled, not run on hardware
//
#include <hip/hip_runtime.h>
#include <hip/hip_bf16.h>

// ---------------------------------------------------------------------------
// Problem constants (from reference): B=64, T=512, N=64, K=2, S=2048, E=32, M=5
// ---------------------------------------------------------------------------
#define BT_TOT 32768   // B*T
#define SK     2048    // S (GEMM K dim)
#define CN     256     // concatenated hyper columns: 160 (w1) + 32 (b1) + 32 (wf) + 32 (v1)
#define NAG    64      // agents
#define MNB    5       // neighborhood size
#define EE     32      // embed dim

// GEMM tiling
#define BM  128
#define BN  64
#define KT  32
#define LDA 40   // LDS row stride (halfs) for A, 80B (16B aligned, conflict-skewed)
#define LDB 40   // LDS row stride (halfs) for B^T ([n][k])

typedef __attribute__((ext_vector_type(16))) _Float16 v16h;
typedef __attribute__((ext_vector_type(8)))  _Float16 v8h;
typedef __attribute__((ext_vector_type(4)))  _Float16 v4h;
typedef __attribute__((ext_vector_type(8)))  float    v8f;

union V16 { v16h v; v8h h[2]; };

// ---------------------------------------------------------------------------
// Kernel 1: pack [W_w1 | W_b1 | W_wf | W_v1] into f16, transposed to [n][k]
// (n = 0..255 concatenated column, k = 0..2047), plus concatenated f32 bias.
// ---------------------------------------------------------------------------
__global__ __launch_bounds__(256)
void qmix_pack(const float* __restrict__ W_w1, const float* __restrict__ b_w1,
               const float* __restrict__ W_b1, const float* __restrict__ b_b1,
               const float* __restrict__ W_wf, const float* __restrict__ b_wf,
               const float* __restrict__ W_v1, const float* __restrict__ b_v1,
               _Float16* __restrict__ Wt, float* __restrict__ biasc) {
  int idx = blockIdx.x * 256 + threadIdx.x;   // 0 .. CN*SK-1
  int n = idx >> 11;                          // column 0..255
  int k = idx & (SK - 1);                     // 0..2047
  float w;
  if (n < 160)      w = W_w1[(size_t)k * 160 + n];
  else if (n < 192) w = W_b1[(size_t)k * 32 + (n - 160)];
  else if (n < 224) w = W_wf[(size_t)k * 32 + (n - 192)];
  else              w = W_v1[(size_t)k * 32 + (n - 224)];
  Wt[(size_t)n * SK + k] = (_Float16)w;
  if (k == 0) {
    float bb;
    if (n < 160)      bb = b_w1[n];
    else if (n < 192) bb = b_b1[n - 160];
    else if (n < 224) bb = b_wf[n - 192];
    else              bb = b_v1[n - 224];
    biasc[n] = bb;
  }
}

// ---------------------------------------------------------------------------
// Kernel 2: hyper = states[BT,2048] @ Wcat[2048,256] + bias   (f16 WMMA, f32 acc)
// Block: 256 threads = 8 waves; block tile 128x64; wave tile 32x32 (2x2 WMMA).
// ---------------------------------------------------------------------------
__global__ __launch_bounds__(256)
void qmix_hyper_gemm(const float* __restrict__ states,
                     const _Float16* __restrict__ Wt,
                     const float* __restrict__ biasc,
                     float* __restrict__ hyper) {
  __shared__ __align__(16) _Float16 As[BM * LDA];
  __shared__ __align__(16) _Float16 Bs[BN * LDB];

  const int tid  = threadIdx.x;
  const int lane = tid & 31;
  const int wave = tid >> 5;
  const int wm = (wave >> 1) * 32;   // wave row offset in block tile (0,32,64,96)
  const int wn = (wave & 1) * 32;    // wave col offset in block tile (0,32)
  const int rowBase = blockIdx.y * BM;
  const int colBase = blockIdx.x * BN;

  const int l16 = lane & 15;
  const int klo = (lane < 16) ? 0 : 8;    // A-matrix K base within 16-group (ISA layout)
  const int kb  = (lane < 16) ? 0 : 8;    // B-matrix K base (ISA layout)

  v8f c[2][2];
#pragma unroll
  for (int i = 0; i < 2; ++i)
#pragma unroll
    for (int j = 0; j < 2; ++j) c[i][j] = (v8f)0.0f;

  for (int kk = 0; kk < SK; kk += KT) {
    // ---- stage A chunk: 128 rows x 32 f32 -> f16 LDS (each thread: 4x float4)
#pragma unroll
    for (int i = 0; i < 4; ++i) {
      int chunk = tid + i * 256;          // 0..1023 float4 chunks
      int row   = chunk >> 3;             // 0..127
      int part  = chunk & 7;              // float4 within row
      const float4 f = *(const float4*)(states + (size_t)(rowBase + row) * SK + kk + part * 4);
      v4h h;
      h[0] = (_Float16)f.x; h[1] = (_Float16)f.y;
      h[2] = (_Float16)f.z; h[3] = (_Float16)f.w;
      *(v4h*)(&As[row * LDA + part * 4]) = h;
    }
    // ---- stage B chunk: 64 cols(n) x 32 halfs(k), already f16 [n][k] in Wt
    {
      int n    = tid >> 2;                // 0..63
      int part = tid & 3;                 // 8 halfs each
      const uint4 w = *(const uint4*)(Wt + (size_t)(colBase + n) * SK + kk + part * 8);
      *(uint4*)(&Bs[n * LDB + part * 8]) = w;
    }
    __syncthreads();

    // ---- fragments (layouts per CDNA5 ISA, wave32)
    V16 a[2], b[2];
#pragma unroll
    for (int mt = 0; mt < 2; ++mt) {
      int row = wm + mt * 16 + l16;
      a[mt].h[0] = *(const v8h*)(&As[row * LDA + klo]);        // K = klo..klo+7
      a[mt].h[1] = *(const v8h*)(&As[row * LDA + 16 + klo]);   // K = 16+klo..16+klo+7
    }
#pragma unroll
    for (int nt = 0; nt < 2; ++nt) {
      int n = wn + nt * 16 + l16;
      b[nt].h[0] = *(const v8h*)(&Bs[n * LDB + kb]);           // K = kb..kb+7
      b[nt].h[1] = *(const v8h*)(&Bs[n * LDB + 16 + kb]);      // K = 16+kb..16+kb+7
    }

#pragma unroll
    for (int mt = 0; mt < 2; ++mt)
#pragma unroll
      for (int nt = 0; nt < 2; ++nt)
        c[mt][nt] = __builtin_amdgcn_wmma_f32_16x16x32_f16(
            /*neg_a=*/false, a[mt].v, /*neg_b=*/false, b[nt].v,
            /*c_mod=*/(short)0, c[mt][nt], /*reuse_a=*/false, /*reuse_b=*/false);

    __syncthreads();
  }

  // ---- epilogue: fused bias add, store f32 hyper
#pragma unroll
  for (int nt = 0; nt < 2; ++nt) {
    int n = colBase + wn + nt * 16 + l16;
    float bv = biasc[n];
#pragma unroll
    for (int mt = 0; mt < 2; ++mt) {
      int r0 = rowBase + wm + mt * 16 + ((lane < 16) ? 0 : 8);
#pragma unroll
      for (int r = 0; r < 8; ++r)
        hyper[(size_t)(r0 + r) * CN + n] = c[mt][nt][r] + bv;
    }
  }
}

// ---------------------------------------------------------------------------
// Kernel 3: per-(b,t) mixing. One 64-thread block (2 waves) per (b,t).
// hyper row: [0..159]=w1 raw (M,E), [160..191]=b1, [192..223]=wf raw, [224..255]=v1 raw
// ---------------------------------------------------------------------------
__global__ __launch_bounds__(64)
void qmix_mixer(const float* __restrict__ hyper,
                const float* __restrict__ agent_qs,
                const int*   __restrict__ nbr,
                const float* __restrict__ W_v2,
                const float* __restrict__ b_v2,
                float* __restrict__ out) {
  const int bt  = blockIdx.x;
  const int tid = threadIdx.x;   // 0..63, one agent per thread

  __shared__ float hrow[CN];
  __shared__ float aq[NAG];
  __shared__ float vsh;

  const float* hp = hyper + (size_t)bt * CN;
#pragma unroll
  for (int i = 0; i < CN / NAG; ++i) hrow[tid + i * NAG] = hp[tid + i * NAG];
  aq[tid] = agent_qs[(size_t)bt * NAG + tid];
  __syncthreads();

  if (tid == 0) {
    float v = b_v2[0];
#pragma unroll
    for (int e = 0; e < EE; ++e) {
      float x = hrow[224 + e];
      v += (x > 0.0f ? x : 0.0f) * W_v2[e];
    }
    vsh = v;
  }
  __syncthreads();

  float u[MNB];
  const int* nb = nbr + tid * MNB;
#pragma unroll
  for (int m = 0; m < MNB; ++m) u[m] = aq[nb[m]];

  float q = vsh;
#pragma unroll 4
  for (int e = 0; e < EE; ++e) {
    float acc = hrow[160 + e];                       // b1[e]
#pragma unroll
    for (int m = 0; m < MNB; ++m)
      acc = fmaf(u[m], fabsf(hrow[m * EE + e]), acc); // u . |w1[:,e]|
    float h = (acc > 0.0f) ? acc : (expf(acc) - 1.0f); // elu
    q = fmaf(h, fabsf(hrow[192 + e]), q);              // h . |wf|
  }
  out[(size_t)bt * NAG + tid] = q;
}

// ---------------------------------------------------------------------------
// Launch
// ---------------------------------------------------------------------------
extern "C" void kernel_launch(void* const* d_in, const int* in_sizes, int n_in,
                              void* d_out, int out_size, void* d_ws, size_t ws_size,
                              hipStream_t stream) {
  const float* agent_qs = (const float*)d_in[0];
  const float* states   = (const float*)d_in[1];
  const int*   nbr      = (const int*)  d_in[2];
  const float* W_w1 = (const float*)d_in[3];
  const float* b_w1 = (const float*)d_in[4];
  const float* W_b1 = (const float*)d_in[5];
  const float* b_b1 = (const float*)d_in[6];
  const float* W_wf = (const float*)d_in[7];
  const float* b_wf = (const float*)d_in[8];
  const float* W_v1 = (const float*)d_in[9];
  const float* b_v1 = (const float*)d_in[10];
  const float* W_v2 = (const float*)d_in[11];
  const float* b_v2 = (const float*)d_in[12];

  // workspace layout
  char* ws = (char*)d_ws;
  _Float16* Wt    = (_Float16*)ws;                              // 256*2048*2 = 1 MB
  float*    biasc = (float*)(ws + (size_t)CN * SK * 2);         // 1 KB
  float*    hyper = (float*)(ws + (size_t)CN * SK * 2 + 4096);  // 32768*256*4 = 32 MB

  qmix_pack<<<(CN * SK) / 256, 256, 0, stream>>>(
      W_w1, b_w1, W_b1, b_b1, W_wf, b_wf, W_v1, b_v1, Wt, biasc);

  qmix_hyper_gemm<<<dim3(CN / BN, BT_TOT / BM), 256, 0, stream>>>(
      states, Wt, biasc, hyper);

  qmix_mixer<<<BT_TOT, 64, 0, stream>>>(
      hyper, agent_qs, nbr, W_v2, b_v2, (float*)d_out);
}
